// FEATTransformer_32040456029043
// MI455X (gfx1250) — matrix operand order repaired and checked by reading, not compiled, not run on hardware
//
#include <hip/hip_runtime.h>

// ---------------------------------------------------------------------------
// CDNA5 (gfx1250, wave32) WMMA bf16 transformer block for MI455X.
// All six GEMMs are compute-bound (103 GFLOP vs ~250MB traffic) -> route
// through v_wmma_f32_16x16x32_bf16 with f32 accumulation.
// ---------------------------------------------------------------------------

typedef __attribute__((ext_vector_type(16))) __bf16 bf16x16;
typedef __attribute__((ext_vector_type(8)))  float  f32x8;

#define BM 128
#define BN 128
#define BK 32
#define KST 40   // LDS row stride in halves (40*2=80B) -> conflict-free b128 reads

__device__ __forceinline__ unsigned short f2bf(float f) {
  unsigned int u = __builtin_bit_cast(unsigned int, f);
  u += 0x7FFFu + ((u >> 16) & 1u);          // round-to-nearest-even
  return (unsigned short)(u >> 16);
}

union FragU {
  bf16x16 v;
  uint4   q[2];
};

// --------------------------- f32 -> bf16 cast ------------------------------
__global__ __launch_bounds__(256)
void cvt_f32_bf16(const float* __restrict__ in, unsigned short* __restrict__ out,
                  long long n) {
  long long i = (long long)blockIdx.x * blockDim.x + threadIdx.x;
  long long stride = (long long)gridDim.x * blockDim.x;
  for (; i < n; i += stride) out[i] = f2bf(in[i]);
}

// --------------------------- C = A * B^T -----------------------------------
// A: [M x K] bf16 row-major, B: [N x K] bf16 row-major, C: [M x N].
// 256 threads = 8 waves; block tile 128x128; wave tile 32x64 (2x4 WMMA frags).
template <bool OUT_BF16>
__global__ __launch_bounds__(256)
void gemm_abt(const unsigned short* __restrict__ A,
              const unsigned short* __restrict__ B,
              void* __restrict__ Cout, int M, int N, int K) {
  __shared__ __align__(16) unsigned short sA[BM * KST];
  __shared__ __align__(16) unsigned short sB[BN * KST];

  const int t     = threadIdx.x;
  const int lane  = t & 31;
  const int wave  = t >> 5;
  const int wm    = wave >> 1;        // 0..3  (row group of 32)
  const int wn    = wave & 1;         // 0..1  (col group of 64)
  const int lrow  = lane & 15;
  const int lhalf = lane >> 4;        // 0 or 1

  const int bm = blockIdx.y * BM;
  const int bn = blockIdx.x * BN;

  // cooperative tile-load mapping: each thread moves 32B of A and 32B of B
  const int cr = t >> 1;              // tile row 0..127
  const int cc = (t & 1) * 16;        // half-offset 0 or 16

  f32x8 acc[2][4];
#pragma unroll
  for (int i = 0; i < 2; ++i)
#pragma unroll
    for (int j = 0; j < 4; ++j) acc[i][j] = (f32x8)0.0f;

  const int kbA = lhalf * 8;          // A frag: chunks at kbA and kbA+16 (ISA layout)
  const int kbB = lhalf * 8;          // B frag: chunks at kbB and kbB+16 (ISA layout)

  for (int k0 = 0; k0 < K; k0 += BK) {
    {
      const uint4* ga = (const uint4*)(A + (size_t)(bm + cr) * K + k0 + cc);
      const uint4* gb = (const uint4*)(B + (size_t)(bn + cr) * K + k0 + cc);
      uint4 a0 = ga[0], a1 = ga[1];
      uint4 b0 = gb[0], b1 = gb[1];
      *(uint4*)&sA[cr * KST + cc]     = a0;
      *(uint4*)&sA[cr * KST + cc + 8] = a1;
      *(uint4*)&sB[cr * KST + cc]     = b0;
      *(uint4*)&sB[cr * KST + cc + 8] = b1;
    }
    __syncthreads();

    FragU a[2], b[4];
#pragma unroll
    for (int mi = 0; mi < 2; ++mi) {
      int row = wm * 32 + mi * 16 + lrow;
      a[mi].q[0] = *(const uint4*)&sA[row * KST + kbA];
      a[mi].q[1] = *(const uint4*)&sA[row * KST + 16 + kbA];
    }
#pragma unroll
    for (int ni = 0; ni < 4; ++ni) {
      int col = wn * 64 + ni * 16 + lrow;
      b[ni].q[0] = *(const uint4*)&sB[col * KST + kbB];
      b[ni].q[1] = *(const uint4*)&sB[col * KST + kbB + 16];
    }
#pragma unroll
    for (int mi = 0; mi < 2; ++mi)
#pragma unroll
      for (int ni = 0; ni < 4; ++ni)
        acc[mi][ni] = __builtin_amdgcn_wmma_f32_16x16x32_bf16(
            false, a[mi].v, false, b[ni].v, (short)0, acc[mi][ni], false, false);
    __syncthreads();
  }

  // C/D layout: VGPR g -> row g + 8*lhalf, col = lane&15
#pragma unroll
  for (int mi = 0; mi < 2; ++mi) {
#pragma unroll
    for (int ni = 0; ni < 4; ++ni) {
#pragma unroll
      for (int g = 0; g < 8; ++g) {
        int row = bm + wm * 32 + mi * 16 + g + 8 * lhalf;
        int col = bn + wn * 64 + ni * 16 + lrow;
        float v = acc[mi][ni][g];
        if (OUT_BF16)
          ((unsigned short*)Cout)[(size_t)row * N + col] = f2bf(v);
        else
          ((float*)Cout)[(size_t)row * N + col] = v;
      }
    }
  }
}

// --------------------------- row softmax -----------------------------------
// attn[row][:] = softmax(scale * S[row][:]) -> bf16
__global__ __launch_bounds__(256)
void softmax_rows(const float* __restrict__ S, unsigned short* __restrict__ attn,
                  int N, float scale) {
  __shared__ float red[256];
  const int   row = blockIdx.x;
  const int   t   = threadIdx.x;
  const float* s  = S + (size_t)row * N;

  float m = -3.402823e38f;
  for (int j = t; j < N; j += 256) m = fmaxf(m, s[j] * scale);
  red[t] = m; __syncthreads();
  for (int o = 128; o > 0; o >>= 1) {
    if (t < o) red[t] = fmaxf(red[t], red[t + o]);
    __syncthreads();
  }
  m = red[0]; __syncthreads();

  float sum = 0.f;
  for (int j = t; j < N; j += 256) sum += __expf(s[j] * scale - m);
  red[t] = sum; __syncthreads();
  for (int o = 128; o > 0; o >>= 1) {
    if (t < o) red[t] += red[t + o];
    __syncthreads();
  }
  const float inv = 1.0f / red[0];

  unsigned short* a = attn + (size_t)row * N;
  for (int j = t; j < N; j += 256) a[j] = f2bf(__expf(s[j] * scale - m) * inv);
}

// ----------------- bias + residual + LayerNorm -----------------------------
__global__ __launch_bounds__(256)
void residual_ln(const float* __restrict__ fcout, const float* __restrict__ fcb,
                 const float* __restrict__ x, const float* __restrict__ gamma,
                 const float* __restrict__ beta, float* __restrict__ out,
                 int d, float eps) {
  __shared__ float r1[256];
  __shared__ float r2[256];
  const int row = blockIdx.x;
  const int t   = threadIdx.x;
  const float* f  = fcout + (size_t)row * d;
  const float* xr = x + (size_t)row * d;

  float s = 0.f, s2 = 0.f;
  for (int j = t; j < d; j += 256) {
    float h = f[j] + fcb[j] + xr[j];
    s += h; s2 += h * h;
  }
  r1[t] = s; r2[t] = s2; __syncthreads();
  for (int o = 128; o > 0; o >>= 1) {
    if (t < o) { r1[t] += r1[t + o]; r2[t] += r2[t + o]; }
    __syncthreads();
  }
  const float mu  = r1[0] / d;
  const float var = r2[0] / d - mu * mu;
  const float rs  = rsqrtf(var + eps);

  float* o = out + (size_t)row * d;
  for (int j = t; j < d; j += 256) {
    float h = f[j] + fcb[j] + xr[j];
    o[j] = (h - mu) * rs * gamma[j] + beta[j];
  }
}

// ---------------------------------------------------------------------------
extern "C" void kernel_launch(void* const* d_in, const int* in_sizes, int n_in,
                              void* d_out, int out_size, void* d_ws, size_t ws_size,
                              hipStream_t stream) {
  const float* x  = (const float*)d_in[0];
  const float* Wq = (const float*)d_in[1];
  const float* Wk = (const float*)d_in[2];
  const float* Wv = (const float*)d_in[3];
  const float* fw = (const float*)d_in[4];
  const float* fb = (const float*)d_in[5];
  const float* gm = (const float*)d_in[6];
  const float* bt = (const float*)d_in[7];

  const int d = in_sizes[5];          // 1024
  const int M = in_sizes[0] / d;      // 4096

  char*  ws  = (char*)d_ws;
  size_t off = 0;
  auto alloc = [&](size_t bytes) -> void* {
    void* p = ws + off;
    off += (bytes + 255) & ~(size_t)255;
    return p;
  };

  unsigned short* xb   = (unsigned short*)alloc((size_t)M * d * 2);
  unsigned short* wqb  = (unsigned short*)alloc((size_t)d * d * 2);
  unsigned short* wkb  = (unsigned short*)alloc((size_t)d * d * 2);
  unsigned short* wvb  = (unsigned short*)alloc((size_t)d * d * 2);
  unsigned short* fwb  = (unsigned short*)alloc((size_t)d * d * 2);
  unsigned short* qb   = (unsigned short*)alloc((size_t)M * d * 2);
  unsigned short* kb   = (unsigned short*)alloc((size_t)M * d * 2);
  unsigned short* vTb  = (unsigned short*)alloc((size_t)d * M * 2);   // V^T [d x M]
  float*          Smat = (float*)         alloc((size_t)M * M * 4);
  unsigned short* attn = (unsigned short*)alloc((size_t)M * M * 2);
  unsigned short* xab  = (unsigned short*)alloc((size_t)M * d * 2);
  float*          fco  = (float*)         alloc((size_t)M * d * 4);

  // 1) f32 -> bf16 casts
  cvt_f32_bf16<<<2048, 256, 0, stream>>>(x,  xb,  (long long)M * d);
  cvt_f32_bf16<<<1024, 256, 0, stream>>>(Wq, wqb, (long long)d * d);
  cvt_f32_bf16<<<1024, 256, 0, stream>>>(Wk, wkb, (long long)d * d);
  cvt_f32_bf16<<<1024, 256, 0, stream>>>(Wv, wvb, (long long)d * d);
  cvt_f32_bf16<<<1024, 256, 0, stream>>>(fw, fwb, (long long)d * d);

  // 2) projections: Q = x Wq^T, K = x Wk^T   [M x d]
  gemm_abt<true><<<dim3(d / BN, M / BM), 256, 0, stream>>>(xb, wqb, qb, M, d, d);
  gemm_abt<true><<<dim3(d / BN, M / BM), 256, 0, stream>>>(xb, wkb, kb, M, d, d);
  //    V^T = Wv x^T  -> same A*B^T kernel, output already transposed [d x M]
  gemm_abt<true><<<dim3(M / BN, d / BM), 256, 0, stream>>>(wvb, xb, vTb, d, M, d);

  // 3) scores S = K Q^T  [M x M]  (scale folded into softmax)
  gemm_abt<false><<<dim3(M / BN, M / BM), 256, 0, stream>>>(kb, qb, Smat, M, M, d);

  // 4) attn = softmax(S / sqrt(d), axis=1) -> bf16
  softmax_rows<<<M, 256, 0, stream>>>(Smat, attn, M, 1.0f / sqrtf((float)d));

  // 5) x_adapted = attn @ V = attn (V^T)^T   [M x d]
  gemm_abt<true><<<dim3(d / BN, M / BM), 256, 0, stream>>>(attn, vTb, xab, M, d, M);

  // 6) fc: out = x_adapted fc_w^T  (bias added in LN kernel)
  gemm_abt<false><<<dim3(d / BN, M / BM), 256, 0, stream>>>(xab, fwb, fco, M, d, d);

  // 7) h = out + fc_b + x ; LayerNorm -> d_out (f32)
  residual_ln<<<M, 256, 0, stream>>>(fco, fb, x, gm, bt, (float*)d_out, d, 1e-5f);
}
